// MHA_81595788689810
// MI455X (gfx1250) — matrix operand order repaired and checked by reading, not compiled, not run on hardware
//
#include <hip/hip_runtime.h>
#include <hip/hip_bf16.h>

#define BB 2
#define SS 2048
#define DD 1024
#define HH 16
#define HDIM 64
#define BHN (BB*HH)
#define BSN (BB*SS)

typedef __attribute__((ext_vector_type(16))) _Float16 v16h;
typedef __attribute__((ext_vector_type(8)))  _Float16 v8h;
typedef __attribute__((ext_vector_type(8)))  float    v8f;
typedef __attribute__((ext_vector_type(4)))  float    v4f;

__device__ __forceinline__ v8f wmma32(v16h a, v16h b, v8f c) {
  // D = A(16x32 f16) * B(32x16 f16) + C(16x16 f32)
  return __builtin_amdgcn_wmma_f32_16x16x32_f16(false, a, false, b, (short)0, c,
                                                false, false);
}

// ---- A fragment (16x32, f16): lane&15 = M row.
// lanes 0-15: K 0..15 contiguous;  lanes 16-31: K 16..31 contiguous.
__device__ __forceinline__ v16h fragA_f16(const _Float16* base, int ld, int lane) {
  int r  = lane & 15;
  int ko = (lane & 16) ? 16 : 0;
  const _Float16* p = base + (size_t)r * ld + ko;
  v8h lo = *(const v8h*)p;
  v8h hi = *(const v8h*)(p + 8);
  return __builtin_shufflevector(lo, hi, 0,1,2,3,4,5,6,7,8,9,10,11,12,13,14,15);
}

__device__ __forceinline__ v16h fragA_f32(const float* base, int ld, int lane) {
  int r  = lane & 15;
  int ko = (lane & 16) ? 16 : 0;
  const float* p = base + (size_t)r * ld + ko;
  v4f a0 = *(const v4f*)(p);
  v4f a1 = *(const v4f*)(p + 4);
  v4f b0 = *(const v4f*)(p + 8);
  v4f b1 = *(const v4f*)(p + 12);
  v16h o;
#pragma unroll
  for (int j = 0; j < 4; ++j) {
    o[j]      = (_Float16)a0[j];
    o[4 + j]  = (_Float16)a1[j];
    o[8 + j]  = (_Float16)b0[j];
    o[12 + j] = (_Float16)b1[j];
  }
  return o;
}

// ---- B fragment (32x16, f16): lane&15 = N column.
// lanes 0-15 hold K=0..15 contiguous; lanes 16-31 hold K=16..31 contiguous.
__device__ __forceinline__ v16h fragB_f16(const _Float16* base, int ld, int lane) {
  int n  = lane & 15;
  int kb = (lane & 16) ? 16 : 0;
  const _Float16* p = base + (size_t)n * ld + kb;
  v8h lo = *(const v8h*)p;
  v8h hi = *(const v8h*)(p + 8);
  return __builtin_shufflevector(lo, hi, 0,1,2,3,4,5,6,7,8,9,10,11,12,13,14,15);
}

__device__ __forceinline__ v16h fragB_f32(const float* base, int ld, int lane) {
  int n  = lane & 15;
  int kb = (lane & 16) ? 16 : 0;
  const float* p = base + (size_t)n * ld + kb;
  v4f x0 = *(const v4f*)(p);
  v4f x1 = *(const v4f*)(p + 4);
  v4f x2 = *(const v4f*)(p + 8);
  v4f x3 = *(const v4f*)(p + 12);
  v16h o;
#pragma unroll
  for (int j = 0; j < 4; ++j) {
    o[j]      = (_Float16)x0[j];
    o[4 + j]  = (_Float16)x1[j];
    o[8 + j]  = (_Float16)x2[j];
    o[12 + j] = (_Float16)x3[j];
  }
  return o;
}

// exchange a v8h with the other 16-lane half (lane ^ 16)
__device__ __forceinline__ v8h xor16_v8h(v8h x) {
  union { v8h h; int i[4]; } u;
  u.h = x;
#pragma unroll
  for (int j = 0; j < 4; ++j) u.i[j] = __shfl_xor(u.i[j], 16, 32);
  return u.h;
}

// ---------------- Projection: Y = X @ W^T + bias, stored f16 --------------
// transposed==0 : Y laid out [B,H,S,HD]   (k)
// transposed==1 : Y laid out [B,H,HD,S]   (v transposed)
__global__ void proj_kernel(const float* __restrict__ X, const float* __restrict__ W,
                            const float* __restrict__ bias, _Float16* __restrict__ Y,
                            int transposed) {
  const int lane = threadIdx.x & 31;
  const int m0 = blockIdx.x * 16;   // row tile (token)
  const int n0 = blockIdx.y * 64;   // col tile base (4 x 16)
  v8f acc[4] = {};
  for (int kk = 0; kk < DD; kk += 32) {
    v16h a = fragA_f32(X + (size_t)m0 * DD + kk, DD, lane);
#pragma unroll
    for (int nt = 0; nt < 4; ++nt) {
      v16h bf = fragB_f32(W + (size_t)(n0 + nt * 16) * DD + kk, DD, lane);
      acc[nt] = wmma32(a, bf, acc[nt]);
    }
  }
  const int col15 = lane & 15;
  const int rbase = (lane & 16) ? 8 : 0;
#pragma unroll
  for (int nt = 0; nt < 4; ++nt) {
    int n = n0 + nt * 16 + col15;
    float bvv = bias[n];
    int h = n >> 6, hd = n & (HDIM - 1);
    if (!transposed) {
#pragma unroll
      for (int r = 0; r < 8; ++r) {
        int row = m0 + rbase + r;
        int b = row >> 11, s = row & (SS - 1);
        Y[((size_t)(b * HH + h) * SS + s) * HDIM + hd] = (_Float16)(acc[nt][r] + bvv);
      }
    } else {
      int row0 = m0 + rbase;
      int b = row0 >> 11, s0 = row0 & (SS - 1);
      v8h pk;
#pragma unroll
      for (int r = 0; r < 8; ++r) pk[r] = (_Float16)(acc[nt][r] + bvv);
      *(v8h*)&Y[((size_t)(b * HH + h) * HDIM + hd) * SS + s0] = pk;
    }
  }
}

// ------------- Pass 1: softmax row stats only (no attn writes) ------------
// Computes S^T tiles = K_tile(16xHD) x Q^T(HDx16); lane owns one query column,
// so max/sum-exp are per-lane scalars; halves combined with one shfl_xor(16).
__global__ void stats_kernel(const _Float16* __restrict__ qh,
                             const _Float16* __restrict__ kh,
                             const int* __restrict__ am,
                             float* __restrict__ rowm, float* __restrict__ rowsum) {
  const int lane = threadIdx.x & 31;
  const int qt = blockIdx.x;
  const int bh = blockIdx.y;
  const int b  = bh / HH;
  const int m0 = qt * 16;
  const int half16 = (lane >> 4) & 1;
  const int query = m0 + (lane & 15);
  const float NEG = -__builtin_inff();

  v16h bq0 = fragB_f16(qh + ((size_t)bh * SS + m0) * HDIM + 0,  HDIM, lane);
  v16h bq1 = fragB_f16(qh + ((size_t)bh * SS + m0) * HDIM + 32, HDIM, lane);

  float m = NEG, s = 0.f;
  for (int kt = 0; kt <= qt; ++kt) {
    v16h ak0 = fragA_f16(kh + ((size_t)bh * SS + kt * 16) * HDIM + 0,  HDIM, lane);
    v16h ak1 = fragA_f16(kh + ((size_t)bh * SS + kt * 16) * HDIM + 32, HDIM, lane);
    v8f sc = {};
    sc = wmma32(ak0, bq0, sc);
    sc = wmma32(ak1, bq1, sc);
    float v[8];
    float tm = NEG;
#pragma unroll
    for (int r = 0; r < 8; ++r) {
      int key = kt * 16 + r + 8 * half16;
      float x = sc[r] * 0.125f;                       // 1/sqrt(HD)
      if (key > query || am[b * SS + key] == 0) x = NEG;
      v[r] = x;
      tm = fmaxf(tm, x);
    }
    float nm = fmaxf(m, tm);
    if (nm > NEG) {                                   // per-lane, no shuffles inside
      float e = 0.f;
#pragma unroll
      for (int r = 0; r < 8; ++r) e += __expf(v[r] - nm);
      s = s * ((m > NEG) ? __expf(m - nm) : 0.f) + e;
      m = nm;
    }
  }
  // combine the two half-wave key partitions
  float mo = __shfl_xor(m, 16, 32);
  float so = __shfl_xor(s, 16, 32);
  float M = fmaxf(m, mo);
  float stot = ((m  > NEG) ? __expf(m  - M) : 0.f) * s +
               ((mo > NEG) ? __expf(mo - M) : 0.f) * so;
  if (lane < 16) {
    size_t idx = (size_t)bh * SS + m0 + lane;
    rowm[idx]   = M;
    rowsum[idx] = stot;
  }
}

// -- Pass 2: recompute scores, normalize, single attn write, fused PV ------
// out^T tiles = vt(HDxS) x P^T ; P^T tiles come straight from the scores
// C-layout registers (one half-wave exchange builds the K=32 B-fragment).
__global__ void attn_pv_kernel(const _Float16* __restrict__ qh,
                               const _Float16* __restrict__ kh,
                               const _Float16* __restrict__ vt,
                               const int* __restrict__ am,
                               const float* __restrict__ rowm,
                               const float* __restrict__ rowsum,
                               float* __restrict__ attn,
                               _Float16* __restrict__ ctx) {
  const int lane = threadIdx.x & 31;
  const int qt = blockIdx.x;
  const int bh = blockIdx.y;
  const int b = bh / HH, h = bh % HH;
  const int m0 = qt * 16;
  const int half16 = (lane >> 4) & 1;
  const int query = m0 + (lane & 15);

  const float mr = rowm[(size_t)bh * SS + query];
  const float sr = rowsum[(size_t)bh * SS + query];
  const float inv = (sr > 0.f) ? (1.f / sr) : 0.f;

  v16h bq0 = fragB_f16(qh + ((size_t)bh * SS + m0) * HDIM + 0,  HDIM, lane);
  v16h bq1 = fragB_f16(qh + ((size_t)bh * SS + m0) * HDIM + 32, HDIM, lane);

  float* arow = attn + ((size_t)bh * SS + query) * SS;   // this lane's query row
  const int valid_end = (qt + 1) * 16;
  const int nc = (qt + 2) >> 1;                          // 32-key chunks
  v8f acc[4] = {};                                       // out^T, hd tiles

  for (int c = 0; c < nc; ++c) {
    v8h ph[2];
#pragma unroll
    for (int pt = 0; pt < 2; ++pt) {
      const int kt = 2 * c + pt;
      if (kt <= qt) {                                    // uniform branch
        v16h ak0 = fragA_f16(kh + ((size_t)bh * SS + kt * 16) * HDIM + 0,  HDIM, lane);
        v16h ak1 = fragA_f16(kh + ((size_t)bh * SS + kt * 16) * HDIM + 32, HDIM, lane);
        v8f sc = {};
        sc = wmma32(ak0, bq0, sc);
        sc = wmma32(ak1, bq1, sc);
        float pf[8];
#pragma unroll
        for (int r = 0; r < 8; ++r) {
          int key = kt * 16 + r + 8 * half16;
          float p = 0.f;
          if (key <= query && am[b * SS + key] != 0)
            p = __expf(sc[r] * 0.125f - mr) * inv;
          pf[r] = p;
          ph[pt][r] = (_Float16)p;
        }
        // normalized attention: contiguous 8-float run per lane
        v4f s0 = {pf[0], pf[1], pf[2], pf[3]};
        v4f s1 = {pf[4], pf[5], pf[6], pf[7]};
        float* pw = arow + kt * 16 + 8 * half16;
        *(v4f*)(pw)     = s0;
        *(v4f*)(pw + 4) = s1;
      } else {
#pragma unroll
        for (int r = 0; r < 8; ++r) ph[pt][r] = (_Float16)0.f;
      }
    }
    // assemble P^T B-fragment (K=32): lanes<16 need keys {0..15} of chunk,
    // lanes>=16 need keys {16..31}; other half's 8 keys come via shfl_xor(16).
    v8h p0x = xor16_v8h(ph[0]);
    v8h p1x = xor16_v8h(ph[1]);
    v16h pb;
#pragma unroll
    for (int j = 0; j < 8; ++j) {
      pb[j]     = half16 ? p1x[j] : ph[0][j];
      pb[8 + j] = half16 ? ph[1][j] : p0x[j];
    }
#pragma unroll
    for (int nt = 0; nt < 4; ++nt) {
      v16h av = fragA_f16(vt + ((size_t)bh * HDIM + nt * 16) * SS + c * 32, SS, lane);
      acc[nt] = wmma32(av, pb, acc[nt]);
    }
  }

  // zero-fill causally-masked upper region (contiguous per query row)
  v4f z = {0.f, 0.f, 0.f, 0.f};
  for (int k = valid_end + 8 * half16; k < SS; k += 16) {
    *(v4f*)(arow + k)     = z;
    *(v4f*)(arow + k + 4) = z;
  }

  // ctx store: lane owns query row, hd contiguous -> one v8h store per tile
#pragma unroll
  for (int nt = 0; nt < 4; ++nt) {
    v8h pk;
#pragma unroll
    for (int r = 0; r < 8; ++r) pk[r] = (_Float16)acc[nt][r];
    int hd = nt * 16 + 8 * half16;
    *(v8h*)&ctx[((size_t)b * SS + query) * DD + h * HDIM + hd] = pk;
  }
}

// ---------------- Output projection: out = ctx @ wo^T + bo (f32) ----------
__global__ void oproj_kernel(const _Float16* __restrict__ ctx,
                             const float* __restrict__ W,
                             const float* __restrict__ bias,
                             float* __restrict__ out) {
  const int lane = threadIdx.x & 31;
  const int m0 = blockIdx.x * 16;
  const int n0 = blockIdx.y * 64;
  v8f acc[4] = {};
  for (int kk = 0; kk < DD; kk += 32) {
    v16h a = fragA_f16(ctx + (size_t)m0 * DD + kk, DD, lane);
#pragma unroll
    for (int nt = 0; nt < 4; ++nt) {
      v16h bf = fragB_f32(W + (size_t)(n0 + nt * 16) * DD + kk, DD, lane);
      acc[nt] = wmma32(a, bf, acc[nt]);
    }
  }
  const int col15 = lane & 15;
  const int rbase = (lane & 16) ? 8 : 0;
#pragma unroll
  for (int nt = 0; nt < 4; ++nt) {
    int n = n0 + nt * 16 + col15;
    float bvv = bias[n];
#pragma unroll
    for (int r = 0; r < 8; ++r) {
      out[(size_t)(m0 + rbase + r) * DD + n] = acc[nt][r] + bvv;
    }
  }
}

extern "C" void kernel_launch(void* const* d_in, const int* in_sizes, int n_in,
                              void* d_out, int out_size, void* d_ws, size_t ws_size,
                              hipStream_t stream) {
  const float* Q  = (const float*)d_in[0];
  const float* K  = (const float*)d_in[1];
  const float* V  = (const float*)d_in[2];
  const int*   am = (const int*)d_in[3];
  const float* wq = (const float*)d_in[4];
  const float* bq = (const float*)d_in[5];
  const float* wk = (const float*)d_in[6];
  const float* bk = (const float*)d_in[7];
  const float* wv = (const float*)d_in[8];
  const float* bv = (const float*)d_in[9];
  const float* wo = (const float*)d_in[10];
  const float* bo = (const float*)d_in[11];

  float* out  = (float*)d_out;
  float* attn = out + (size_t)BSN * DD;   // outputs: [out | attn], flat

  char* w = (char*)d_ws;
  _Float16* qh  = (_Float16*)w; w += (size_t)BSN * DD * 2;  // [B,H,S,HD]
  _Float16* kh  = (_Float16*)w; w += (size_t)BSN * DD * 2;  // [B,H,S,HD]
  _Float16* vt  = (_Float16*)w; w += (size_t)BSN * DD * 2;  // [B,H,HD,S]
  _Float16* ctx = (_Float16*)w; w += (size_t)BSN * DD * 2;  // [B,S,D]
  float* rowm   = (float*)w;    w += (size_t)BHN * SS * 4;
  float* rowsum = (float*)w;

  dim3 tb(32);                               // one wave32 per block
  dim3 pg(BSN / 16, DD / 64);                // 256 x 16 tiles
  proj_kernel<<<pg, tb, 0, stream>>>(Q, wq, bq, qh, 0);
  proj_kernel<<<pg, tb, 0, stream>>>(K, wk, bk, kh, 0);
  proj_kernel<<<pg, tb, 0, stream>>>(V, wv, bv, vt, 1);

  dim3 ag(SS / 16, BHN);                     // 128 x 32
  stats_kernel<<<ag, tb, 0, stream>>>(qh, kh, am, rowm, rowsum);
  attn_pv_kernel<<<ag, tb, 0, stream>>>(qh, kh, vt, am, rowm, rowsum, attn, ctx);

  oproj_kernel<<<pg, tb, 0, stream>>>(ctx, wo, bo, out);
}
